// RelationModule_50233937494175
// MI455X (gfx1250) — matrix operand order repaired and checked by reading, not compiled, not run on hardware
//
#include <hip/hip_runtime.h>
#include <hip/hip_bf16.h>

// ---------------------------------------------------------------------------
// RelationModule for MI455X (gfx1250, wave32, WMMA bf16 16x16x32)
// 4-way register-blocked GEMMs: one A fragment reused across 4 B fragments.
// ---------------------------------------------------------------------------

typedef __attribute__((ext_vector_type(16))) __bf16          v16bf;
typedef __attribute__((ext_vector_type(8)))  float           v8f;
typedef __attribute__((ext_vector_type(8)))  unsigned short  ushort8;
typedef __attribute__((ext_vector_type(4)))  unsigned short  ushort4v;
typedef __attribute__((ext_vector_type(4)))  float           float4v;

union FragAB { ushort8 u8[2]; v16bf v; };

constexpr int N_  = 1024;
constexpr int D_  = 2048;
constexpr int H_  = 16;
constexpr int DK_ = 128;
constexpr int DG_ = 128;
constexpr int HD_ = H_ * DK_;   // 2048
constexpr int PE_LD = 136;      // 128 + 8 shorts pad -> 272B row stride (17x16B, no bank conflicts)

__device__ __forceinline__ unsigned short f2bf(float x) {
  union { float f; unsigned u; } a; a.f = x;
  unsigned u = a.u;
  u += 0x7FFFu + ((u >> 16) & 1u);        // round-to-nearest-even
  return (unsigned short)(u >> 16);
}

__device__ __forceinline__ v8f wmma_bf16(const FragAB& a, const FragAB& b, v8f c) {
  return __builtin_amdgcn_wmma_f32_16x16x32_bf16(false, a.v, false, b.v,
                                                 (short)0, c, false, false);
}

// ---------------------------------------------------------------------------
// fp32 -> bf16 bulk convert
__global__ void cvt_bf16_kernel(const float* __restrict__ src,
                                unsigned short* __restrict__ dst, int n) {
  int i = blockIdx.x * blockDim.x + threadIdx.x;
  if (i < n) dst[i] = f2bf(src[i]);
}

// ---------------------------------------------------------------------------
// per-box geometry: cx, cy, w, h (clipped)
__global__ void geom_kernel(const float* __restrict__ bbox, float* __restrict__ g) {
  int n = blockIdx.x * blockDim.x + threadIdx.x;
  if (n >= N_) return;
  const float* b = bbox + n * 5;          // (1,N,5), drop col 0
  float xmin = b[1], ymin = b[2], xmax = b[3], ymax = b[4];
  g[n]          = 0.5f * (xmin + xmax);
  g[N_ + n]     = 0.5f * (ymin + ymax);
  g[2 * N_ + n] = fmaxf(xmax - xmin, 1e-4f);
  g[3 * N_ + n] = fmaxf(ymax - ymin, 1e-4f);
}

// ---------------------------------------------------------------------------
// C(1024x2048) = Xb(1024x2048) @ W^T, W given as (out=2048, in=2048) bf16 = B^T
// Each wave: 16 rows x 64 cols (4 accumulator tiles, A fragment reused 4x).
// mode 0: Q  -> out[n][col] = bf16((acc+bias)*scale)   (scale = 1/sqrt(DK))
// mode 1: K  -> out[n][col]
// mode 2: V  -> out[col][n]  (transposed store for attn@V B-fragments)
__global__ __launch_bounds__(256)
void proj_gemm_kernel(const unsigned short* __restrict__ Xb,
                      const unsigned short* __restrict__ Wb,
                      const float* __restrict__ bias,
                      unsigned short* __restrict__ out,
                      int mode, float scale) {
  const int lane = threadIdx.x & 31, wave = threadIdx.x >> 5;
  const int half = lane >> 4, lr = lane & 15;
  const int ctg = blockIdx.x * 8 + wave;       // 0..31, 4 col-tiles each
  const int nt  = blockIdx.y;                  // 0..63 row tile
  const int n   = nt * 16 + lr;
  const unsigned short* arow = Xb + (size_t)n * D_;
  const unsigned short* brow[4];
  int col[4];
#pragma unroll
  for (int j = 0; j < 4; ++j) {
    col[j]  = (ctg * 4 + j) * 16 + lr;
    brow[j] = Wb + (size_t)col[j] * D_;
  }
  v8f acc[4] = {};
#pragma unroll 2
  for (int kt = 0; kt < D_ / 32; ++kt) {
    const int k0 = kt * 32;
    FragAB a;
    a.u8[0] = *(const ushort8*)(arow + k0 + half * 8);
    a.u8[1] = *(const ushort8*)(arow + k0 + 16 + half * 8);
#pragma unroll
    for (int j = 0; j < 4; ++j) {
      FragAB b;
      b.u8[0] = *(const ushort8*)(brow[j] + k0 + half * 8);
      b.u8[1] = *(const ushort8*)(brow[j] + k0 + 16 + half * 8);
      acc[j] = wmma_bf16(a, b, acc[j]);
    }
  }
#pragma unroll
  for (int j = 0; j < 4; ++j) {
    const float bv = bias[col[j]];
    if (mode == 2) {                           // V^T store: 8 contiguous bf16
      ushort8 pack;
#pragma unroll
      for (int i = 0; i < 8; ++i) pack[i] = f2bf((acc[j][i] + bv) * scale);
      *(ushort8*)(out + (size_t)col[j] * N_ + nt * 16 + half * 8) = pack;
    } else {
#pragma unroll
      for (int i = 0; i < 8; ++i) {
        int nn = nt * 16 + half * 8 + i;
        out[(size_t)nn * HD_ + col[j]] = f2bf((acc[j][i] + bv) * scale);
      }
    }
  }
}

// ---------------------------------------------------------------------------
// Geometric bias: S[h][n][m] = relu( pe(n,m,:) . WG[h,:] + WG_b[h] )
// pe tile (256 m x 128 g) built in LDS as bf16, consumed by WMMA (heads = N dim)
__global__ __launch_bounds__(256)
void wg_pe_kernel(const float* __restrict__ geom,
                  const unsigned short* __restrict__ WGb,
                  const float* __restrict__ WG_b,
                  float* __restrict__ S) {
  __shared__ __align__(16) unsigned short pe[256 * PE_LD];
  const int t = threadIdx.x;
  const int n = blockIdx.y;
  const int mbase = blockIdx.x * 256;

  const float cxn = geom[n], cyn = geom[N_ + n];
  const float wn = geom[2 * N_ + n], hn = geom[3 * N_ + n];
  const float lwn = __logf(wn), lhn = __logf(hn);

  const int m = mbase + t;
  const float cxm = geom[m], cym = geom[N_ + m];
  const float wm = geom[2 * N_ + m], hm = geom[3 * N_ + m];

  float pos[4];
  pos[0] = __logf(fmaxf(fabsf((cxn - cxm) / wn), 1e-3f));
  pos[1] = __logf(fmaxf(fabsf((cyn - cym) / hn), 1e-3f));
  pos[2] = lwn - __logf(wm);
  pos[3] = lhn - __logf(hm);

  unsigned short* row = pe + t * PE_LD;
#pragma unroll
  for (int c = 0; c < 4; ++c) {
    const float base = 100.f * pos[c];
#pragma unroll
    for (int f = 0; f < 16; ++f) {
      // dim_mat[f] = 1000^(-f/16) = exp(-ln(1000)*f/16)
      const float dm = __expf(-6.907755278982137f * ((float)f * (1.f / 16.f)));
      float s, co;
      __sincosf(base * dm, &s, &co);
      row[c * 16 + f]      = f2bf(s);
      row[64 + c * 16 + f] = f2bf(co);
    }
  }
  __syncthreads();

  const int lane = t & 31, wave = t >> 5;
  const int half = lane >> 4, lr = lane & 15;
  // B fragment: B[g][h], B^T row h = WG[h][:] contiguous
  FragAB bfr[4];
#pragma unroll
  for (int kt = 0; kt < 4; ++kt) {
    const unsigned short* bp = WGb + lr * DG_ + kt * 32 + half * 8;
    bfr[kt].u8[0] = *(const ushort8*)(bp);
    bfr[kt].u8[1] = *(const ushort8*)(bp + 16);
  }
  const float hb = WG_b[lr];
#pragma unroll
  for (int rep = 0; rep < 2; ++rep) {
    const int tile = wave * 2 + rep;                 // 0..15 (16 m each)
    const unsigned short* arow = pe + (size_t)(tile * 16 + lr) * PE_LD;
    v8f acc = {};
#pragma unroll
    for (int kt = 0; kt < 4; ++kt) {
      FragAB a;
      a.u8[0] = *(const ushort8*)(arow + kt * 32 + half * 8);
      a.u8[1] = *(const ushort8*)(arow + kt * 32 + 16 + half * 8);
      acc = wmma_bf16(a, bfr[kt], acc);
    }
    // lane: head = lr; element i -> m = mbase + tile*16 + 8*half + i (contiguous)
    float* sp = S + ((size_t)lr * N_ + n) * N_ + mbase + tile * 16 + half * 8;
#pragma unroll
    for (int i = 0; i < 8; ++i) sp[i] = fmaxf(acc[i] + hb, 0.f);
  }
}

// ---------------------------------------------------------------------------
// S[h][nt][mt] = wg (preloaded as C) + Q_scaled @ K^T   (Q already * 1/sqrt(DK))
// Each wave: one n-tile x 4 m-tiles; Q fragment reused 4x per K step.
__global__ __launch_bounds__(256)
void scores_kernel(const unsigned short* __restrict__ Qb,
                   const unsigned short* __restrict__ Kb,
                   float* __restrict__ S) {
  const int lane = threadIdx.x & 31, wave = threadIdx.x >> 5;
  const int half = lane >> 4, lr = lane & 15;
  const int mtg = blockIdx.x * 8 + wave;  // 0..15, 4 m-tiles each
  const int nt  = blockIdx.y;             // 0..63
  const int h   = blockIdx.z;             // 0..15
  const unsigned short* arow = Qb + (size_t)(nt * 16 + lr) * HD_ + h * DK_;
  const unsigned short* brow[4];
  float* sbase[4];
  v8f acc[4];
#pragma unroll
  for (int j = 0; j < 4; ++j) {
    const int mt = mtg * 4 + j;
    brow[j]  = Kb + (size_t)(mt * 16 + lr) * HD_ + h * DK_;
    sbase[j] = S + ((size_t)h * N_ + nt * 16) * N_ + mt * 16;
#pragma unroll
    for (int i = 0; i < 8; ++i)
      acc[j][i] = sbase[j][(size_t)(8 * half + i) * N_ + lr];
  }
#pragma unroll
  for (int kt = 0; kt < 4; ++kt) {
    FragAB a;
    a.u8[0] = *(const ushort8*)(arow + kt * 32 + half * 8);
    a.u8[1] = *(const ushort8*)(arow + kt * 32 + 16 + half * 8);
#pragma unroll
    for (int j = 0; j < 4; ++j) {
      FragAB b;
      b.u8[0] = *(const ushort8*)(brow[j] + kt * 32 + half * 8);
      b.u8[1] = *(const ushort8*)(brow[j] + kt * 32 + 16 + half * 8);
      acc[j] = wmma_bf16(a, b, acc[j]);
    }
  }
#pragma unroll
  for (int j = 0; j < 4; ++j)
#pragma unroll
    for (int i = 0; i < 8; ++i)
      sbase[j][(size_t)(8 * half + i) * N_ + lr] = acc[j][i];
}

// ---------------------------------------------------------------------------
// row softmax over m (1024), write attn as bf16
__global__ __launch_bounds__(256)
void softmax_kernel(const float* __restrict__ S, unsigned short* __restrict__ P) {
  const int row = blockIdx.x;                       // h*N + n
  const float* s = S + (size_t)row * N_;
  __shared__ float red[256];
  const int t = threadIdx.x;
  float4v v = *(const float4v*)(s + t * 4);
  float mx = fmaxf(fmaxf(v[0], v[1]), fmaxf(v[2], v[3]));
  red[t] = mx; __syncthreads();
  for (int off = 128; off > 0; off >>= 1) {
    if (t < off) red[t] = fmaxf(red[t], red[t + off]);
    __syncthreads();
  }
  mx = red[0]; __syncthreads();
  float e0 = __expf(v[0] - mx), e1 = __expf(v[1] - mx);
  float e2 = __expf(v[2] - mx), e3 = __expf(v[3] - mx);
  red[t] = e0 + e1 + e2 + e3; __syncthreads();
  for (int off = 128; off > 0; off >>= 1) {
    if (t < off) red[t] += red[t + off];
    __syncthreads();
  }
  const float inv = 1.0f / red[0];
  ushort4v o;
  o[0] = f2bf(e0 * inv); o[1] = f2bf(e1 * inv);
  o[2] = f2bf(e2 * inv); o[3] = f2bf(e3 * inv);
  *(ushort4v*)(P + (size_t)row * N_ + t * 4) = o;
}

// ---------------------------------------------------------------------------
// out[n][h*DK+k] = sum_m P[h][n][m] * V[h][m][k]   (B^T = V^T stored row-major)
// Each wave: 16 rows x 64 of DK (4 k-tiles); attn fragment reused 4x per step.
__global__ __launch_bounds__(256)
void out_gemm_kernel(const unsigned short* __restrict__ P,
                     const unsigned short* __restrict__ VTb,
                     float* __restrict__ out) {
  const int lane = threadIdx.x & 31, wave = threadIdx.x >> 5;
  const int half = lane >> 4, lr = lane & 15;
  const int ntl = wave >> 1;              // 0..3
  const int ktg = (wave & 1) * 4;         // 0 or 4 (4 k-tiles each)
  const int nt  = blockIdx.x * 4 + ntl;   // 0..63
  const int h   = blockIdx.y;             // 0..15
  const unsigned short* arow = P + ((size_t)h * N_ + nt * 16 + lr) * N_;
  const unsigned short* brow[4];
#pragma unroll
  for (int j = 0; j < 4; ++j)
    brow[j] = VTb + (size_t)(h * DK_ + (ktg + j) * 16 + lr) * N_;
  v8f acc[4] = {};
#pragma unroll 2
  for (int ms = 0; ms < N_ / 32; ++ms) {
    FragAB a;
    a.u8[0] = *(const ushort8*)(arow + ms * 32 + half * 8);
    a.u8[1] = *(const ushort8*)(arow + ms * 32 + 16 + half * 8);
#pragma unroll
    for (int j = 0; j < 4; ++j) {
      FragAB b;
      b.u8[0] = *(const ushort8*)(brow[j] + ms * 32 + half * 8);
      b.u8[1] = *(const ushort8*)(brow[j] + ms * 32 + 16 + half * 8);
      acc[j] = wmma_bf16(a, b, acc[j]);
    }
  }
#pragma unroll
  for (int j = 0; j < 4; ++j) {
    const int colj = h * DK_ + (ktg + j) * 16 + lr;
#pragma unroll
    for (int i = 0; i < 8; ++i)
      out[(size_t)(nt * 16 + 8 * half + i) * HD_ + colj] = acc[j][i];
  }
}

// ---------------------------------------------------------------------------
extern "C" void kernel_launch(void* const* d_in, const int* in_sizes, int n_in,
                              void* d_out, int out_size, void* d_ws, size_t ws_size,
                              hipStream_t stream) {
  (void)in_sizes; (void)n_in; (void)out_size; (void)ws_size;
  const float* app   = (const float*)d_in[0];
  const float* bbox  = (const float*)d_in[1];
  const float* WQ_w  = (const float*)d_in[2];
  const float* WQ_b  = (const float*)d_in[3];
  const float* WK_w  = (const float*)d_in[4];
  const float* WK_b  = (const float*)d_in[5];
  const float* WV_w  = (const float*)d_in[6];
  const float* WV_b  = (const float*)d_in[7];
  const float* WG_w  = (const float*)d_in[8];
  const float* WG_b  = (const float*)d_in[9];
  float* out = (float*)d_out;

  // workspace layout (bytes, 256-aligned sections)
  char* ws = (char*)d_ws;
  size_t off = 0;
  auto alloc = [&](size_t bytes) { char* p = ws + off; off = (off + bytes + 255) & ~(size_t)255; return p; };
  unsigned short* Xb  = (unsigned short*)alloc((size_t)N_ * D_ * 2);     // 4 MB
  unsigned short* Wqb = (unsigned short*)alloc((size_t)HD_ * D_ * 2);    // 8 MB
  unsigned short* Wkb = (unsigned short*)alloc((size_t)HD_ * D_ * 2);    // 8 MB
  unsigned short* Wvb = (unsigned short*)alloc((size_t)HD_ * D_ * 2);    // 8 MB
  unsigned short* WGb = (unsigned short*)alloc((size_t)H_ * DG_ * 2);    // 4 KB
  float*          geo = (float*)alloc((size_t)4 * N_ * 4);               // 16 KB
  unsigned short* Qb  = (unsigned short*)alloc((size_t)N_ * HD_ * 2);    // 4 MB
  unsigned short* Kb  = (unsigned short*)alloc((size_t)N_ * HD_ * 2);    // 4 MB
  unsigned short* VTb = (unsigned short*)alloc((size_t)HD_ * N_ * 2);    // 4 MB
  float*          S   = (float*)alloc((size_t)H_ * N_ * N_ * 4);         // 64 MB
  unsigned short* P   = (unsigned short*)alloc((size_t)H_ * N_ * N_ * 2);// 32 MB

  // 1) conversions + geometry
  const int XN = N_ * D_, WN = H_ * DK_ * D_, GN = H_ * DG_;
  cvt_bf16_kernel<<<(XN + 255) / 256, 256, 0, stream>>>(app,  Xb,  XN);
  cvt_bf16_kernel<<<(WN + 255) / 256, 256, 0, stream>>>(WQ_w, Wqb, WN);
  cvt_bf16_kernel<<<(WN + 255) / 256, 256, 0, stream>>>(WK_w, Wkb, WN);
  cvt_bf16_kernel<<<(WN + 255) / 256, 256, 0, stream>>>(WV_w, Wvb, WN);
  cvt_bf16_kernel<<<(GN + 255) / 256, 256, 0, stream>>>(WG_w, WGb, GN);
  geom_kernel<<<(N_ + 255) / 256, 256, 0, stream>>>(bbox, geo);

  // 2) projections: Q (pre-scaled by 1/sqrt(DK)), K, V^T
  const float qscale = 0.08838834764831845f;   // 1/sqrt(128)
  dim3 pgrid(4, 64);
  proj_gemm_kernel<<<pgrid, 256, 0, stream>>>(Xb, Wqb, WQ_b, Qb,  0, qscale);
  proj_gemm_kernel<<<pgrid, 256, 0, stream>>>(Xb, Wkb, WK_b, Kb,  1, 1.0f);
  proj_gemm_kernel<<<pgrid, 256, 0, stream>>>(Xb, Wvb, WV_b, VTb, 2, 1.0f);

  // 3) geometric bias into S (consumed as WMMA C accumulator by scores)
  wg_pe_kernel<<<dim3(N_ / 256, N_), 256, 0, stream>>>(geo, WGb, WG_b, S);

  // 4) S += Q K^T (scaled)
  scores_kernel<<<dim3(2, 64, H_), 256, 0, stream>>>(Qb, Kb, S);

  // 5) softmax rows -> bf16 attn
  softmax_kernel<<<H_ * N_, 256, 0, stream>>>(S, P);

  // 6) out = attn @ V, concat heads
  out_gemm_kernel<<<dim3(16, H_), 256, 0, stream>>>(P, VTb, out);
}
